// CT_ADRN_42314017800796
// MI455X (gfx1250) — matrix operand order repaired and checked by reading, not compiled, not run on hardware
//
#include <hip/hip_runtime.h>
#include <hip/hip_bf16.h>
#include <cmath>

// ---------------- vector types (gfx1250, wave32) ----------------
typedef __attribute__((ext_vector_type(16))) _Float16 v16h;
typedef __attribute__((ext_vector_type(8)))  float    v8f;
typedef __attribute__((ext_vector_type(4)))  float    f32x4;
typedef __attribute__((ext_vector_type(4)))  unsigned int u32x4;
typedef __attribute__((ext_vector_type(8)))  int      i32x8;
typedef __attribute__((ext_vector_type(4)))  int      i32x4;

__device__ __forceinline__ float siluf(float v){ return v / (1.f + __expf(-v)); }

// A-fragment K permutation for V_WMMA_F32_16X16X32_F16 (ISA 7.12.2, 16-bit A 16x32):
// lanes 0-15 consume K [0..7,16..23], lanes 16-31 consume K [8..15,24..31].
__device__ __forceinline__ int kpermA(int s){
  int q = s >> 3;
  return (q == 1) ? (s + 8) : (q == 2) ? (s - 8) : s;
}

// Load the A fragment from an LDS tile stored in *natural* f32 [row][32] layout:
// per-lane permuted gather (two 8-element runs) + cvt to f16.
__device__ __forceinline__ v16h load_a_frag_f32(const float* arow, int hh)
{
  f32x4 a0 = *(const f32x4*)(arow + hh*8);
  f32x4 a1 = *(const f32x4*)(arow + hh*8 + 4);
  f32x4 a2 = *(const f32x4*)(arow + 16 + hh*8);
  f32x4 a3 = *(const f32x4*)(arow + 16 + hh*8 + 4);
  v16h a;
  #pragma unroll
  for (int i = 0; i < 4; ++i) {
    a[i]      = (_Float16)a0[i];
    a[4 + i]  = (_Float16)a1[i];
    a[8 + i]  = (_Float16)a2[i];
    a[12 + i] = (_Float16)a3[i];
  }
  return a;
}

// =====================================================================
// Implicit-GEMM convolution on WMMA, with TDM weight staging.
//   D[co, n] = sum_k W[co,k] * im2col[k, n]    n=(b,oy,ox), k=(ci,ky,kx)
// Block tile 64(M=co) x 64(N=pos), K-chunk 32. 256 threads = 8 waves as
// 4(M)x2(N); each wave owns two 16x16 accumulators (A-fragment reuse).
// wmode==0: A tile staged by TENSOR_LOAD_TO_LDS (wave 0 issues the DMA,
//           s_wait_tensorcnt 0, block barrier). Tensor dims are set to the
//           *remaining* rows/cols so TDM OOB handling zero-fills tails.
// wmode==1 (+ldil==2): conv_transpose2d — flipped/transposed weights staged
//           manually, lhs-dilated im2col gather.
// Epilogue fuses conv bias + per-(b,co) time-embedding bias + residual add.
// =====================================================================
__global__ __launch_bounds__(256)
void k_conv_igemm_wmma(const float* __restrict__ X, const float* __restrict__ Wt,
                       const float* __restrict__ bias, const float* __restrict__ badd,
                       const float* __restrict__ res, float* __restrict__ Y,
                       int Cin, int H, int Win,
                       int Cout, int Ho, int Wo,
                       int KH, int KW, int stride, int pad, int ldil, int wmode,
                       int logWo, int logHoWo, unsigned long long kmag, unsigned kwmag,
                       int Ntot)
{
  __shared__ float    As32[64*32] __attribute__((aligned(128)));
  __shared__ _Float16 Bs[64*32]   __attribute__((aligned(64)));

  const int tid = threadIdx.x, lane = tid & 31, wv = tid >> 5;
  const int mi = wv & 3, nj = wv >> 2;
  const int m0 = blockIdx.y * 64, n0 = blockIdx.x * 64;
  const int HoWo = Ho * Wo;
  const int KK = KH * KW;
  const int Ktot = Cin * KK;
  const unsigned ldsA = (unsigned)(unsigned long long)(void*)As32;

  v8f acc0 = {}, acc1 = {};

  for (int ko = 0; ko < Ktot; ko += 32) {
    // ---- stage A (weights) ----
    if (wmode == 0) {
      if (wv == 0) {
        // Tensor DMA descriptor (D#), 2D tensor: rows=C_out slice, cols=K slice
        unsigned long long ga =
            (unsigned long long)(const void*)(Wt + (size_t)m0 * Ktot + ko);
        int d0 = Ktot - ko;                       // remaining cols (K)
        int d1 = Cout - m0;                       // remaining rows (co)
        u32x4 g0;
        g0.x = 1u;                                // count=1, user descriptor
        g0.y = ldsA;                              // lds_addr
        g0.z = (unsigned)ga;                      // global_addr[31:0]
        g0.w = (unsigned)((ga >> 32) & 0x01FFFFFFu) | (2u << 30);  // addr[56:32], type=2
        i32x8 g1;
        g1[0] = 0x00020000;                                               // data_size=4B
        g1[1] = (int)(((unsigned)d0 & 0xFFFFu) << 16);                    // tensor_dim0 lo
        g1[2] = (int)((((unsigned)d0 >> 16) & 0xFFFFu) |
                      (((unsigned)d1 & 0xFFFFu) << 16));                  // dim0 hi | dim1 lo
        g1[3] = (int)((((unsigned)d1 >> 16) & 0xFFFFu) | (32u << 16));    // dim1 hi | tile_dim0=32
        g1[4] = 64;                                                       // tile_dim1=64
        g1[5] = Ktot;                                                     // dim0 stride lo
        g1[6] = 0;
        g1[7] = 0;
        i32x4 z4 = {0, 0, 0, 0};
#if defined(__clang_major__) && (__clang_major__ >= 23)
        i32x8 z8 = {0, 0, 0, 0, 0, 0, 0, 0};
        __builtin_amdgcn_tensor_load_to_lds(g0, g1, z4, z4, z8, 0);
#else
        __builtin_amdgcn_tensor_load_to_lds(g0, g1, z4, z4, 0);
#endif
      }
    } else {
      // convT weights: W[ci, co, KH-1-ky, KW-1-kx], natural K layout in LDS
      for (int idx = tid; idx < 64*32; idx += 256) {
        int m = idx >> 5, slot = idx & 31;
        int kg = ko + slot, co = m0 + m;
        float v = 0.f;
        if (co < Cout && kg < Ktot) {
          int ci = (int)(((unsigned long long)kg * kmag) >> 40);
          int r  = kg - ci * KK;
          int ky = (int)(((unsigned)r * kwmag) >> 16);
          int kx = r - ky * KW;
          v = Wt[(((size_t)ci * Cout + co) * KH + (KH-1-ky)) * KW + (KW-1-kx)];
        }
        As32[idx] = v;
      }
    }
    // ---- stage B (im2col), 64 positions x 32 K, kpermA K order (matches A fragment) ----
    for (int idx = tid; idx < 64*32; idx += 256) {
      int n = idx >> 5, slot = idx & 31;
      int kg = ko + kpermA(slot), np = n0 + n;
      float v = 0.f;
      if (np < Ntot && kg < Ktot) {
        int b  = np >> logHoWo;
        int rr = np & (HoWo - 1);
        int oy = rr >> logWo, ox = rr & (Wo - 1);
        int ci = (int)(((unsigned long long)kg * kmag) >> 40);
        int r2 = kg - ci * KK;
        int ky = (int)(((unsigned)r2 * kwmag) >> 16);
        int kx = r2 - ky * KW;
        int iy = oy * stride - pad + ky;
        int ix = ox * stride - pad + kx;
        bool ok = true;
        if (ldil == 2) {
          if (iy < 0 || ix < 0 || (iy & 1) || (ix & 1)) ok = false;
          iy >>= 1; ix >>= 1;
        }
        if (ok && iy >= 0 && iy < H && ix >= 0 && ix < Win)
          v = X[(((size_t)b * Cin + ci) * H + iy) * Win + ix];
      }
      Bs[idx] = (_Float16)v;
    }
    if (tid == 0 && (ko + 32) < Ktot)
      __builtin_prefetch(&Wt[(size_t)m0 * Ktot + ko + 32], 0, 0);   // global_prefetch_b8
    if (wmode == 0 && wv == 0)
      __builtin_amdgcn_s_wait_tensorcnt(0);
    __syncthreads();

    const int lm = lane & 15, hh = lane >> 4;
    v16h a  = load_a_frag_f32(&As32[(mi*16 + lm) * 32], hh);
    v16h b0 = *(const v16h*)&Bs[(nj*32 + lm) * 32 + hh*16];
    v16h b1 = *(const v16h*)&Bs[(nj*32 + 16 + lm) * 32 + hh*16];
    acc0 = __builtin_amdgcn_wmma_f32_16x16x32_f16(false, a, false, b0,
                                                  (short)0, acc0, false, false);
    acc1 = __builtin_amdgcn_wmma_f32_16x16x32_f16(false, a, false, b1,
                                                  (short)0, acc1, false, false);
    __syncthreads();
  }

  // ---- epilogue: D layout (VGPR r: M=r | r+8; N=lane%16) ----
  const int lm = lane & 15, hi = lane >> 4;
  #pragma unroll
  for (int r = 0; r < 8; ++r) {
    int co = m0 + mi*16 + r + hi*8;
    if (co < Cout) {
      float bv = bias ? bias[co] : 0.f;
      #pragma unroll
      for (int tile = 0; tile < 2; ++tile) {
        int np = n0 + nj*32 + tile*16 + lm;
        if (np < Ntot) {
          int b = np >> logHoWo, rr = np & (HoWo - 1);
          size_t off = ((size_t)b * Cout + co) * (size_t)HoWo + rr;
          float v = (tile ? acc1[r] : acc0[r]) + bv;
          if (badd) v += badd[(size_t)b * Cout + co];
          if (res)  v += res[off];
          Y[off] = v;
        }
      }
    }
  }
}

// =====================================================================
// Generic batched WMMA GEMM (f32 in/out, f16 compute):
//   C[m,n] = alpha * sum_k A(m,k) * B(k,n)
// A(m,k) = A[transA ? k*lda+m : m*lda+k]; batched over grid.z = B*8 heads.
// Block tile 64x64, 2 accumulators per wave.
// =====================================================================
__global__ __launch_bounds__(256)
void k_gemm_wmma(const float* __restrict__ A, const float* __restrict__ B_,
                 float* __restrict__ C,
                 int M, int N, int K, int lda, int ldb, int ldc,
                 int transA, int transB, float alpha,
                 long sAb, long sAh, long sBb, long sBh, long sCb, long sCh)
{
  __shared__ _Float16 As[64*32] __attribute__((aligned(64)));
  __shared__ _Float16 Bs[64*32] __attribute__((aligned(64)));

  const int z = blockIdx.z;
  const float* Ap = A  + (z >> 3) * sAb + (z & 7) * sAh;
  const float* Bp = B_ + (z >> 3) * sBb + (z & 7) * sBh;
  float*       Cp = C  + (z >> 3) * sCb + (z & 7) * sCh;

  const int tid = threadIdx.x, lane = tid & 31, wv = tid >> 5;
  const int mi = wv & 3, nj = wv >> 2;
  const int m0 = blockIdx.y * 64, n0 = blockIdx.x * 64;

  v8f acc0 = {}, acc1 = {};
  for (int ko = 0; ko < K; ko += 32) {
    for (int idx = tid; idx < 64*32; idx += 256) {
      int m = idx >> 5, slot = idx & 31;
      int kg = ko + kpermA(slot); int mg = m0 + m;
      float v = 0.f;
      if (mg < M && kg < K)
        v = Ap[transA ? (size_t)kg * lda + mg : (size_t)mg * lda + kg];
      As[idx] = (_Float16)v;
    }
    for (int idx = tid; idx < 64*32; idx += 256) {
      int n = idx >> 5, slot = idx & 31;
      int kg = ko + kpermA(slot); int ng = n0 + n;
      float v = 0.f;
      if (ng < N && kg < K)
        v = Bp[transB ? (size_t)ng * ldb + kg : (size_t)kg * ldb + ng];
      Bs[idx] = (_Float16)v;
    }
    __syncthreads();
    const int lm = lane & 15, hh = lane >> 4;
    v16h a  = *(const v16h*)&As[(mi*16 + lm) * 32 + hh*16];
    v16h b0 = *(const v16h*)&Bs[(nj*32 + lm) * 32 + hh*16];
    v16h b1 = *(const v16h*)&Bs[(nj*32 + 16 + lm) * 32 + hh*16];
    acc0 = __builtin_amdgcn_wmma_f32_16x16x32_f16(false, a, false, b0,
                                                  (short)0, acc0, false, false);
    acc1 = __builtin_amdgcn_wmma_f32_16x16x32_f16(false, a, false, b1,
                                                  (short)0, acc1, false, false);
    __syncthreads();
  }
  const int lm = lane & 15, hi = lane >> 4;
  #pragma unroll
  for (int r = 0; r < 8; ++r) {
    int mg = m0 + mi*16 + r + hi*8;
    if (mg < M) {
      #pragma unroll
      for (int tile = 0; tile < 2; ++tile) {
        int ng = n0 + nj*32 + tile*16 + lm;
        if (ng < N) Cp[(size_t)mg * ldc + ng] = alpha * (tile ? acc1[r] : acc0[r]);
      }
    }
  }
}

// ---------------- row softmax (in place) ----------------
__global__ __launch_bounds__(256)
void k_softmax_rows(float* __restrict__ S, int n)
{
  __shared__ float red[256];
  float* row = S + (size_t)blockIdx.x * n;
  const int tid = threadIdx.x;
  float mx = -1e30f;
  for (int i = tid; i < n; i += 256) mx = fmaxf(mx, row[i]);
  red[tid] = mx; __syncthreads();
  for (int s = 128; s > 0; s >>= 1){ if (tid < s) red[tid] = fmaxf(red[tid], red[tid+s]); __syncthreads(); }
  mx = red[0]; __syncthreads();
  float sum = 0.f;
  for (int i = tid; i < n; i += 256){ float e = __expf(row[i]-mx); row[i] = e; sum += e; }
  red[tid] = sum; __syncthreads();
  for (int s = 128; s > 0; s >>= 1){ if (tid < s) red[tid] += red[tid+s]; __syncthreads(); }
  const float inv = 1.f / red[0];
  for (int i = tid; i < n; i += 256) row[i] *= inv;
}

// ---------------- group norm (+ optional fused SiLU) ----------------
__global__ __launch_bounds__(256)
void k_groupnorm(const float* __restrict__ X, float* __restrict__ Y,
                 const float* __restrict__ gamma, const float* __restrict__ beta,
                 int C, int HW, int logHW, int groups, int dosilu)
{
  const int bg = blockIdx.x, b = bg / 8, g = bg % 8;
  const int cpg = C / groups;
  const size_t base = ((size_t)b * C + (size_t)g * cpg) * HW;
  const int n = cpg * HW;
  __shared__ float r1[256], r2[256];
  float s = 0.f, ss = 0.f;
  for (int i = threadIdx.x; i < n; i += 256){ float v = X[base+i]; s += v; ss += v*v; }
  r1[threadIdx.x] = s; r2[threadIdx.x] = ss; __syncthreads();
  for (int st = 128; st > 0; st >>= 1){
    if (threadIdx.x < st){ r1[threadIdx.x] += r1[threadIdx.x+st]; r2[threadIdx.x] += r2[threadIdx.x+st]; }
    __syncthreads();
  }
  const float mean = r1[0] / n;
  const float var  = r2[0] / n - mean*mean;
  const float inv  = rsqrtf(var + 1e-5f);
  for (int i = threadIdx.x; i < n; i += 256){
    int c = g*cpg + (i >> logHW);
    float v = (X[base+i] - mean) * inv * gamma[c] + beta[c];
    if (dosilu) v = siluf(v);
    Y[base+i] = v;
  }
}

// ---------------- small dense linear (time MLP / temb projection) ----------------
__global__ __launch_bounds__(256)
void k_linear(const float* __restrict__ X, const float* __restrict__ Wt,
              const float* __restrict__ bias, float* __restrict__ Y,
              int NB, int I, int O, int siluIn)
{
  int idx = blockIdx.x * blockDim.x + threadIdx.x;
  if (idx >= NB * O) return;
  int b = idx / O, o = idx % O;
  float s = bias[o];
  const float* xr = X + (size_t)b * I;
  const float* wr = Wt + (size_t)o * I;
  for (int i = 0; i < I; ++i){ float v = xr[i]; if (siluIn) v = siluf(v); s += v * wr[i]; }
  Y[idx] = s;
}

// ---------------- sinusoidal timestep embedding (B=2, dim=256) ----------------
__global__ void k_timestep(const int* __restrict__ t, float* __restrict__ out)
{
  int idx = blockIdx.x * blockDim.x + threadIdx.x;
  if (idx >= 2*256) return;
  int b = idx / 256, j = idx % 256;
  const int half = 128;
  const float lf = -logf(10000.f) / (half - 1);
  float tv = (float)t[b];
  if (j < half) out[idx] = __sinf(tv * __expf(lf * j));
  else          out[idx] = __cosf(tv * __expf(lf * (j - half)));
}

// =====================================================================
// Host-side orchestration
// =====================================================================
namespace {

struct CPar { const float* b; const float* w; };           // conv / linear / convT
struct GPar { const float* b; const float* g; };           // groupnorm
struct RPar { CPar c1, c2; GPar n1, n2; CPar skip; bool has_skip; CPar tm; };
struct APar { GPar n; CPar proj, qkv; };

struct Cur {
  void* const* d; int i;
  const float* f(){ return (const float*)d[i++]; }
};
static CPar getC(Cur& c){ CPar p; p.b = c.f(); p.w = c.f(); return p; }
static GPar getG(Cur& c){ GPar p; p.b = c.f(); p.g = c.f(); return p; }
static RPar getR(Cur& c, bool skip){
  RPar r{}; r.c1 = getC(c); r.c2 = getC(c); r.n1 = getG(c); r.n2 = getG(c);
  r.has_skip = skip; if (skip) r.skip = getC(c); r.tm = getC(c); return r;
}
static APar getA(Cur& c){ APar a; a.n = getG(c); a.proj = getC(c); a.qkv = getC(c); return a; }

struct WS {
  char* base; size_t off;
  float* alloc(size_t nelem){
    size_t bytes = (nelem * sizeof(float) + 255) & ~(size_t)255;
    float* p = (float*)(base + off); off += bytes; return p;
  }
};

static inline int ilog2i(int v){ int l = 0; while ((1 << l) < v) ++l; return l; }

static void conv(hipStream_t st, const float* X, const CPar& p, float* Y,
                 int Cin, int H, int Cout, int Ho,
                 int KH, int stride, int pad, int ldil, int wmode,
                 const float* badd, const float* resid)
{
  const int KW = KH, KK = KH * KW;
  const int Ntot = 2 * Ho * Ho;
  const int logWo = ilog2i(Ho), logHoWo = ilog2i(Ho * Ho);
  const unsigned long long kmag = ((1ULL << 40) + KK - 1) / (unsigned long long)KK;
  const unsigned kwmag = (65536u + KW - 1) / (unsigned)KW;
  dim3 g((Ntot + 63) / 64, (Cout + 63) / 64, 1);
  k_conv_igemm_wmma<<<g, 256, 0, st>>>(X, p.w, p.b, badd, resid, Y,
      Cin, H, H, Cout, Ho, Ho, KH, KW, stride, pad, ldil, wmode,
      logWo, logHoWo, kmag, kwmag, Ntot);
}

static void gnorm(hipStream_t st, const float* X, float* Y, const GPar& p,
                  int C, int HW, int dosilu)
{
  k_groupnorm<<<2*8, 256, 0, st>>>(X, Y, p.g, p.b, C, HW, ilog2i(HW), 8, dosilu);
}

static void res_block(hipStream_t st, WS& ws, const float* x, float* out,
                      int ci, int co, int Hs, const RPar& p, const float* temb)
{
  size_t mark = ws.off;
  int HW = Hs * Hs;
  float* t0 = ws.alloc((size_t)2*ci*HW);
  gnorm(st, x, t0, p.n1, ci, HW, 1);
  float* tp = ws.alloc((size_t)2*co);
  k_linear<<<(2*co + 255)/256, 256, 0, st>>>(temb, p.tm.w, p.tm.b, tp, 2, 256, co, 1);
  float* t1 = ws.alloc((size_t)2*co*HW);
  conv(st, t0, p.c1, t1, ci, Hs, co, Hs, 3, 1, 1, 1, 0, tp, nullptr);
  float* t2 = ws.alloc((size_t)2*co*HW);
  gnorm(st, t1, t2, p.n2, co, HW, 1);
  const float* resid = x;
  if (p.has_skip){
    float* sk = ws.alloc((size_t)2*co*HW);
    conv(st, x, p.skip, sk, ci, Hs, co, Hs, 1, 1, 0, 1, 0, nullptr, nullptr);
    resid = sk;
  }
  conv(st, t2, p.c2, out, co, Hs, co, Hs, 3, 1, 1, 1, 0, nullptr, resid);
  ws.off = mark;
}

static void attention(hipStream_t st, WS& ws, const float* x, float* out,
                      int C, int Hs, const APar& p, float* scores)
{
  size_t mark = ws.off;
  const int HW = Hs*Hs, hd = C/8, S = HW;
  float* t0 = ws.alloc((size_t)2*C*HW);
  gnorm(st, x, t0, p.n, C, HW, 0);
  float* qkv = ws.alloc((size_t)2*3*C*HW);
  conv(st, t0, p.qkv, qkv, C, Hs, 3*C, Hs, 1, 1, 0, 1, 0, nullptr, nullptr);

  const float scale = 1.0f / sqrtf((float)hd);
  // scores[d,f] = scale * sum_c q[c,d] k[c,f]   (Q^T K)
  dim3 g1((S + 63)/64, (S + 63)/64, 16);
  k_gemm_wmma<<<g1, 256, 0, st>>>(qkv, qkv + (size_t)C*HW, scores,
      S, S, hd, HW, HW, S, 1, 0, scale,
      (long)3*C*HW, (long)hd*HW, (long)3*C*HW, (long)hd*HW,
      (long)8*(long)S*S, (long)S*S);
  k_softmax_rows<<<16*S, 256, 0, st>>>(scores, S);

  // out[c,d] = sum_f v[c,f] attn[d,f]   (V A^T)
  float* aout = ws.alloc((size_t)2*C*HW);
  dim3 g2((S + 63)/64, (hd + 63)/64, 16);
  k_gemm_wmma<<<g2, 256, 0, st>>>(qkv + (size_t)2*C*HW, scores, aout,
      hd, S, S, HW, S, HW, 0, 1, 1.f,
      (long)3*C*HW, (long)hd*HW, (long)8*(long)S*S, (long)S*S,
      (long)C*HW, (long)hd*HW);

  conv(st, aout, p.proj, out, C, Hs, C, Hs, 1, 1, 0, 1, 0, nullptr, x);
  ws.off = mark;
}

static void concat2(hipStream_t st, const float* a, int C1, const float* b, int C2,
                    int HW, float* dst)
{
  for (int n = 0; n < 2; ++n) {
    hipMemcpyAsync(dst + ((size_t)n*(C1+C2))*HW, a + (size_t)n*C1*HW,
                   (size_t)C1*HW*sizeof(float), hipMemcpyDeviceToDevice, st);
    hipMemcpyAsync(dst + ((size_t)n*(C1+C2) + C1)*HW, b + (size_t)n*C2*HW,
                   (size_t)C2*HW*sizeof(float), hipMemcpyDeviceToDevice, st);
  }
}

} // namespace

extern "C" void kernel_launch(void* const* d_in, const int* in_sizes, int n_in,
                              void* d_out, int out_size, void* d_ws, size_t ws_size,
                              hipStream_t stream)
{
  (void)in_sizes; (void)n_in; (void)out_size; (void)ws_size;
  const float* x = (const float*)d_in[0];
  const int*   t = (const int*)d_in[1];

  // ---- consume params in jax pytree (sorted-key) leaf order ----
  Cur c{d_in, 2};
  APar decAttn{}; RPar decR1[4], decR2[4]; CPar decUp[4];
  const bool decHasAttn[4] = {true, false, false, false};
  const int  decUpC[4] = {512, 256, 128, 64};
  const int  decCo[4]  = {256, 128, 64, 64};
  for (int l = 0; l < 4; ++l) {
    if (decHasAttn[l]) decAttn = getA(c);
    decR1[l] = getR(c, true);
    decR2[l] = getR(c, false);
    decUp[l] = getC(c);
  }
  CPar dec_final = getC(c);
  APar encAttn[4]{}; RPar encR1[4], encR2[4]; CPar encPool[4];
  const bool encHasAttn[4] = {false, false, true, true};
  const int  encCi[4] = {64, 64, 128, 256};
  const int  encCo[4] = {64, 128, 256, 512};
  for (int l = 0; l < 4; ++l) {
    if (encHasAttn[l]) encAttn[l] = getA(c);
    encPool[l] = getC(c);
    encR1[l] = getR(c, l > 0);
    encR2[l] = getR(c, false);
  }
  CPar pInit   = getC(c);
  APar midAttn = getA(c);
  RPar midR1   = getR(c, false);
  RPar midR2   = getR(c, false);
  CPar outConv = getC(c);
  GPar outGn   = getG(c);
  CPar tmlp1   = getC(c);
  CPar tmlp2   = getC(c);

  // ---- workspace ----
  WS ws{(char*)d_ws, 0};
  const size_t MAXE = (size_t)2 * 128 * 128 * 128;   // biggest activation (concat @128)
  float* hb0 = ws.alloc(MAXE);
  float* hb1 = ws.alloc(MAXE);
  float* hb2 = ws.alloc(MAXE);
  float* scores = ws.alloc((size_t)2 * 8 * 1024 * 1024);  // largest attn S=1024
  const int skC[4]  = {64, 128, 256, 512};
  const int skHs[4] = {128, 64, 32, 16};
  float* skip[4];
  for (int l = 0; l < 4; ++l) skip[l] = ws.alloc((size_t)2 * skC[l] * skHs[l] * skHs[l]);
  float* te0  = ws.alloc(2*256);
  float* te1  = ws.alloc(2*1024);
  float* temb = ws.alloc(2*256);

  // ---- time embedding MLP ----
  k_timestep<<<2, 256, 0, stream>>>(t, te0);
  k_linear<<<(2*1024 + 255)/256, 256, 0, stream>>>(te0, tmlp1.w, tmlp1.b, te1, 2, 256, 1024, 0);
  k_linear<<<(2*256  + 255)/256, 256, 0, stream>>>(te1, tmlp2.w, tmlp2.b, temb, 2, 1024, 256, 1);

  // ---- init conv: 1 -> 64 @128 ----
  conv(stream, x, pInit, hb0, 1, 128, 64, 128, 3, 1, 1, 1, 0, nullptr, nullptr);

  // ---- encoder ----
  float* curp = hb0;
  int Hs = 128;
  for (int l = 0; l < 4; ++l) {
    int ci = encCi[l], co = encCo[l];
    res_block(stream, ws, curp, hb1, ci, co, Hs, encR1[l], temb);
    if (encHasAttn[l]) {
      res_block(stream, ws, hb1, hb2, co, co, Hs, encR2[l], temb);
      attention(stream, ws, hb2, skip[l], co, Hs, encAttn[l], scores);
    } else {
      res_block(stream, ws, hb1, skip[l], co, co, Hs, encR2[l], temb);
    }
    conv(stream, skip[l], encPool[l], hb0, co, Hs, co, Hs/2, 3, 2, 1, 1, 0, nullptr, nullptr);
    curp = hb0;
    Hs /= 2;
  }

  // ---- mid (C=512 @8) ----
  res_block(stream, ws, curp, hb1, 512, 512, Hs, midR1, temb);
  attention(stream, ws, hb1, hb2, 512, Hs, midAttn, scores);
  res_block(stream, ws, hb2, hb0, 512, 512, Hs, midR2, temb);
  curp = hb0;

  // ---- decoder ----
  for (int l = 0; l < 4; ++l) {
    int uc = decUpC[l], co = decCo[l];
    int Ho = Hs * 2;
    float* tUp = (curp == hb1) ? hb0 : hb1;
    // conv_transpose2d: KH=4, pad=2, lhs_dilation=2, flipped/transposed weights
    conv(stream, curp, decUp[l], tUp, uc, Hs, uc, Ho, 4, 1, 2, 2, 1, nullptr, nullptr);
    concat2(stream, tUp, uc, skip[3-l], skC[3-l], Ho*Ho, hb2);
    res_block(stream, ws, hb2, hb0, 2*uc, co, Ho, decR1[l], temb);
    res_block(stream, ws, hb0, hb1, co, co, Ho, decR2[l], temb);
    if (decHasAttn[l]) { attention(stream, ws, hb1, hb0, co, Ho, decAttn, scores); curp = hb0; }
    else curp = hb1;
    Hs = Ho;
  }

  // ---- head: dec_final -> out_gn+silu -> out_conv -> d_out ----
  float* tF = (curp == hb0) ? hb1 : hb0;
  conv(stream, curp, dec_final, tF, 64, 128, 64, 128, 3, 1, 1, 1, 0, nullptr, nullptr);
  gnorm(stream, tF, hb2, outGn, 64, 128*128, 1);
  conv(stream, hb2, outConv, (float*)d_out, 64, 128, 1, 128, 3, 1, 1, 1, 0, nullptr, nullptr);
}
